// ParallelHyperbolicEncoder_31533649887456
// MI455X (gfx1250) — compile-verified, matrix operand order checked by reading, not run on hardware
//
#include <hip/hip_runtime.h>
#include <hip/hip_bf16.h>
#include <math.h>

typedef __attribute__((ext_vector_type(16))) _Float16 v16h;
typedef __attribute__((ext_vector_type(8)))  float    v8f;

#define BR     4
#define BATCH  32
#define TLEN   1024
#define HDIM   64
#define DIN    128
#define NSTATE 16
#define KCONV  4
#define RRANK  4
#define EDIM   32
#define LNUM   3
#define MROWS  (BATCH*TLEN)          // 32768 rows per branch
#define XDBLW  (RRANK + 2*NSTATE)    // 36
#define CT     16                    // scan LDS staging chunk (timesteps)

#if __has_builtin(__builtin_amdgcn_global_load_async_to_lds_b32) && \
    __has_builtin(__builtin_amdgcn_s_wait_asynccnt)
#define HAVE_ASYNC_LDS 1
#else
#define HAVE_ASYNC_LDS 0
#endif

typedef __attribute__((address_space(1))) int g1int_t;   // global (AS1) int
typedef __attribute__((address_space(3))) int l3int_t;   // LDS (AS3) int

__device__ __forceinline__ float sigmoidf_fast(float x){ return 1.f/(1.f+__expf(-x)); }
__device__ __forceinline__ float siluf(float x){ return x*sigmoidf_fast(x); }

enum class ASrc { Plain, YSiluZ };

// ---------------------------------------------------------------------------
// Register-blocked WMMA GEMM: C[M,N] = act( A[M,K] @ W[N,K]^T (+bias) ).
// One wave computes a 16(M) x (16*NBLK)(N) strip: the A operand is loaded once
// per K-step and reused across NBLK v_wmma_f32_16x16x32_f16 issues.
// A operand lane packing (ISA 7.12.2, 16-bit A 16x32): lane m (m<16) holds
// K = kk+0..7 (elems 0..7) and kk+16..23 (elems 8..15); lane m+16 holds
// K = kk+8..15 and kk+24..31. B operand uses symmetric packing on W[N,K].
// ---------------------------------------------------------------------------
template<int N, int K, int NBLK, ASrc AS, bool BIAS_TANH>
__global__ __launch_bounds__(256)
void wmma_gemm(const float* __restrict__ A, long aBr,
               const float* __restrict__ Z, long zBr, int zLd, int zCol,
               const float* __restrict__ W, long wBr,
               const float* __restrict__ bias, long biasBr,
               float* __restrict__ C, long cBr)
{
  constexpr int NT   = (N + 15) / 16;
  constexpr int NGRP = (NT + NBLK - 1) / NBLK;
  constexpr int MT   = MROWS / 16;
  const int br = blockIdx.z;
  const float* Ab = A + (long)br * aBr;
  const float* Wb = W + (long)br * wBr;
  float*       Cb = C + (long)br * cBr;

  const int wave = blockIdx.x * (blockDim.x >> 5) + (threadIdx.x >> 5);
  if (wave >= MT * NGRP) return;               // wave-uniform exit
  const int mt   = wave / NGRP;
  const int ng   = wave % NGRP;
  const int lane = threadIdx.x & 31;
  const int l15  = lane & 15;
  const int hi   = lane >> 4;

  const int arow = mt * 16 + l15;

  v8f acc[NBLK];
  const v8f vzero = {};
  #pragma unroll
  for (int s = 0; s < NBLK; ++s) acc[s] = vzero;

  #pragma unroll
  for (int kk = 0; kk < K; kk += 32) {
    const int o0 = kk + (hi ? 8  : 0);
    const int o1 = kk + (hi ? 24 : 16);

    v16h a;
    if constexpr (AS == ASrc::Plain) {
      const float* p0 = Ab + (long)arow * K + o0;
      const float* p1 = Ab + (long)arow * K + o1;
      if (kk + 32 < K) __builtin_prefetch(p0 + 32, 0, 1);
      #pragma unroll
      for (int j = 0; j < 8; ++j) a[j]     = (_Float16)p0[j];
      #pragma unroll
      for (int j = 0; j < 8; ++j) a[8 + j] = (_Float16)p1[j];
    } else {                                   // fused A = y * silu(z)
      const float* py0 = Ab + (long)arow * K + o0;
      const float* py1 = Ab + (long)arow * K + o1;
      const float* pz0 = Z + (long)br * zBr + (long)arow * zLd + zCol + o0;
      const float* pz1 = Z + (long)br * zBr + (long)arow * zLd + zCol + o1;
      #pragma unroll
      for (int j = 0; j < 8; ++j) a[j]     = (_Float16)(py0[j] * siluf(pz0[j]));
      #pragma unroll
      for (int j = 0; j < 8; ++j) a[8 + j] = (_Float16)(py1[j] * siluf(pz1[j]));
    }

    #pragma unroll
    for (int s = 0; s < NBLK; ++s) {
      const int brow = (ng * NBLK + s) * 16 + l15;
      v16h b;
      if (brow < N) {
        const float* q0 = Wb + (long)brow * K + o0;
        const float* q1 = Wb + (long)brow * K + o1;
        #pragma unroll
        for (int j = 0; j < 8; ++j) b[j]     = (_Float16)q0[j];
        #pragma unroll
        for (int j = 0; j < 8; ++j) b[8 + j] = (_Float16)q1[j];
      } else {
        #pragma unroll
        for (int j = 0; j < 16; ++j) b[j] = (_Float16)0.f;
      }
      acc[s] = __builtin_amdgcn_wmma_f32_16x16x32_f16(
          /*neg_a=*/false, a, /*neg_b=*/false, b,
          /*c_mod=*/(short)0, acc[s], /*reuse_a=*/false, /*reuse_b=*/false);
    }
  }

  #pragma unroll
  for (int s = 0; s < NBLK; ++s) {
    const int brow = (ng * NBLK + s) * 16 + l15;
    if (brow < N) {
      float bv = 0.f;
      if constexpr (BIAS_TANH) bv = (bias + (long)br * biasBr)[brow];
      #pragma unroll
      for (int i = 0; i < 8; ++i) {   // C/D: lane<16 -> M=i, else M=8+i; N=lane%16
        const int m = mt * 16 + (hi ? 8 + i : i);
        float v = acc[s][i];
        if constexpr (BIAS_TANH) v = tanhf(v + bv);
        Cb[(long)m * N + brow] = v;
      }
    }
  }
}

// ---------------------------------------------------------------------------
// h0 = x * inp_W + inp_b   (scalar -> H=64 embed, all 4 branches)
// ---------------------------------------------------------------------------
__global__ __launch_bounds__(256)
void embed_kernel(const float* __restrict__ x0, const float* __restrict__ x1,
                  const float* __restrict__ x2, const float* __restrict__ x3,
                  const float* __restrict__ iW, const float* __restrict__ ib,
                  float* __restrict__ H)
{
  long idx = (long)blockIdx.x * blockDim.x + threadIdx.x;   // BR*MROWS*HDIM
  if (idx >= (long)BR * MROWS * HDIM) return;
  const int  hh = (int)(idx % HDIM);
  const long r  = idx / HDIM;
  const int  br = (int)(r / MROWS);
  const long m  = r % MROWS;
  const float* xp = (br == 0) ? x0 : (br == 1) ? x1 : (br == 2) ? x2 : x3;
  H[idx] = xp[m] * iW[br * HDIM + hh] + ib[br * HDIM + hh];
}

// ---------------------------------------------------------------------------
// Depthwise causal conv (K=4, left-pad 3) + bias + SiLU on xr (cols 0..127 of xz)
// ---------------------------------------------------------------------------
__global__ __launch_bounds__(256)
void conv_silu_kernel(const float* __restrict__ XZ,
                      const float* __restrict__ cW, long cwBr,
                      const float* __restrict__ cb, long cbBr,
                      float* __restrict__ U)
{
  const int br = blockIdx.z;
  long idx = (long)blockIdx.x * blockDim.x + threadIdx.x;   // MROWS*DIN
  if (idx >= (long)MROWS * DIN) return;
  const int  d = (int)(idx & (DIN - 1));
  const long m = idx >> 7;
  const int  t = (int)(m & (TLEN - 1));
  const float* xz = XZ + (long)br * MROWS * (2 * DIN);
  const float* w  = cW + (long)br * cwBr + d * KCONV;
  float acc = cb[(long)br * cbBr + d];
  #pragma unroll
  for (int j = 0; j < KCONV; ++j) {
    const int tt = t - (KCONV - 1) + j;
    const float xv = (tt >= 0) ? xz[(m + (long)(tt - t)) * (2 * DIN) + d] : 0.f;
    acc = fmaf(xv, w[j], acc);
  }
  U[(long)br * MROWS * DIN + idx] = siluf(acc);
}

// ---------------------------------------------------------------------------
// Selective scan: block = (br,b), 128 threads = Din channels, 16-float state
// in registers. x_dbl rows for CT=16 timesteps staged into LDS per refill
// via global_load_async_to_lds_b32 (ASYNCcnt path) where available.
// ---------------------------------------------------------------------------
__global__ __launch_bounds__(DIN)
void scan_kernel(const float* __restrict__ XD, const float* __restrict__ U,
                 const float* __restrict__ dtW, long dtwBr,
                 const float* __restrict__ dtb, long dtbBr,
                 const float* __restrict__ A_log, long alBr,
                 const float* __restrict__ Dp, long dpBr,
                 float* __restrict__ Y)
{
  const int br = blockIdx.z;
  const int b  = blockIdx.x;
  const int d  = threadIdx.x;
  __shared__ float shxd[CT * XDBLW];          // 576 floats
  const float* xd = XD + (long)br * MROWS * XDBLW + (long)b * TLEN * XDBLW;
  const float* u  = U  + (long)br * MROWS * DIN   + (long)b * TLEN * DIN;
  float*       y  = Y  + (long)br * MROWS * DIN   + (long)b * TLEN * DIN;

  const float* wdt = dtW + (long)br * dtwBr + d * RRANK;
  const float w0 = wdt[0], w1 = wdt[1], w2 = wdt[2], w3 = wdt[3];
  const float bdt = dtb[(long)br * dtbBr + d];
  const float dsk = Dp[(long)br * dpBr + d];

  float a[NSTATE];
  const float* al = A_log + (long)br * alBr + d * NSTATE;
  #pragma unroll
  for (int n = 0; n < NSTATE; ++n) a[n] = -__expf(al[n]);   // A = -exp(A_log)

  float st[NSTATE];
  #pragma unroll
  for (int n = 0; n < NSTATE; ++n) st[n] = 0.f;

  for (int t0 = 0; t0 < TLEN; t0 += CT) {
    __syncthreads();                          // previous chunk fully consumed
    for (int i = threadIdx.x; i < CT * XDBLW; i += DIN) {
      const float* gp = xd + (long)t0 * XDBLW + i;
#if HAVE_ASYNC_LDS
      __builtin_amdgcn_global_load_async_to_lds_b32(
          (g1int_t*)gp, (l3int_t*)&shxd[i], 0, 0);
#else
      shxd[i] = *gp;
#endif
    }
#if HAVE_ASYNC_LDS
    __builtin_amdgcn_s_wait_asynccnt(0);
#endif
    __syncthreads();                          // chunk visible to all waves

    for (int tc = 0; tc < CT; ++tc) {
      const int t = t0 + tc;
      const float* sh = &shxd[tc * XDBLW];
      const float raw = bdt + sh[0]*w0 + sh[1]*w1 + sh[2]*w2 + sh[3]*w3;
      const float dt  = (raw > 20.f) ? raw : log1pf(__expf(raw));   // softplus
      const float ut  = u[(long)t * DIN + d];
      const float dtu = dt * ut;
      float yt = 0.f;
      #pragma unroll
      for (int n = 0; n < NSTATE; ++n) {
        const float dA = __expf(dt * a[n]);
        st[n] = fmaf(dA, st[n], dtu * sh[RRANK + n]);
        yt    = fmaf(st[n], sh[RRANK + NSTATE + n], yt);
      }
      y[(long)t * DIN + d] = fmaf(dsk, ut, yt);
    }
  }
}

// ---------------------------------------------------------------------------
// Hyperbolic combine: one wave32 per (b,t) row, lane = component (E=32).
// ---------------------------------------------------------------------------
__device__ __forceinline__ float wave_sum(float v) {
  #pragma unroll
  for (int off = 16; off > 0; off >>= 1) v += __shfl_xor(v, off, 32);
  return v;
}

__device__ __forceinline__ float mobius_add_lane(float x, float y) {
  const float x2 = wave_sum(x * x);
  const float y2 = wave_sum(y * y);
  const float xy = wave_sum(x * y);
  const float num = (1.f + 2.f * xy + y2) * x + (1.f - x2) * y;
  const float den = fmaxf(1.f + 2.f * xy + x2 * y2, 1e-15f);
  return num / den;
}

__global__ __launch_bounds__(256)
void hyper_kernel(const float* __restrict__ ZE, float* __restrict__ out)
{
  const float MAXN = 1.0f - 0.004f;
  const long row = (long)blockIdx.x * (blockDim.x >> 5) + (threadIdx.x >> 5);
  if (row >= MROWS) return;                      // wave-uniform
  const int lane = threadIdx.x & 31;

  float v[BR];
  #pragma unroll
  for (int brn = 0; brn < BR; ++brn) {
    const float z = ZE[(long)brn * MROWS * EDIM + row * EDIM + lane];
    const float n = sqrtf(fmaxf(wave_sum(z * z), 1e-15f));   // expmap0
    float x = tanhf(n) * z / n;
    const float m = sqrtf(fmaxf(wave_sum(x * x), 1e-15f));   // projx
    if (m > MAXN) x *= MAXN / m;
    v[brn] = x;
  }
  const float wd   = mobius_add_lane(v[1], v[2]);
  const float twd  = mobius_add_lane(v[0], wd);
  float comb       = mobius_add_lane(twd, v[3]);
  const float cm   = sqrtf(fmaxf(wave_sum(comb * comb), 1e-15f));
  if (cm > MAXN) comb *= MAXN / cm;

  #pragma unroll
  for (int s = 0; s < 4; ++s)
    out[(long)s * MROWS * EDIM + row * EDIM + lane] = v[s];
  out[(long)4 * MROWS * EDIM + row * EDIM + lane] = comb;
}

// ---------------------------------------------------------------------------
// Host-side orchestration
// ---------------------------------------------------------------------------
static inline dim3 gemm_grid(int n, int nblk) {
  const int nt   = (n + 15) / 16;
  const int ngrp = (nt + nblk - 1) / nblk;
  const int tiles = (MROWS / 16) * ngrp;
  return dim3((tiles + 7) / 8, 1, BR);
}

extern "C" void kernel_launch(void* const* d_in, const int* in_sizes, int n_in,
                              void* d_out, int out_size, void* d_ws, size_t ws_size,
                              hipStream_t stream) {
  (void)in_sizes; (void)n_in; (void)out_size; (void)ws_size;
  const float* x0   = (const float*)d_in[0];
  const float* x1   = (const float*)d_in[1];
  const float* x2   = (const float*)d_in[2];
  const float* x3   = (const float*)d_in[3];
  const float* inpW = (const float*)d_in[4];
  const float* inpB = (const float*)d_in[5];
  const float* ipW  = (const float*)d_in[6];   // (Br,L,256,64)
  const float* cW   = (const float*)d_in[7];   // (Br,L,128,4)
  const float* cB   = (const float*)d_in[8];   // (Br,L,128)
  const float* xpW  = (const float*)d_in[9];   // (Br,L,36,128)
  const float* dtW  = (const float*)d_in[10];  // (Br,L,128,4)
  const float* dtB  = (const float*)d_in[11];  // (Br,L,128)
  const float* Alog = (const float*)d_in[12];  // (Br,L,128,16)
  const float* Dsk  = (const float*)d_in[13];  // (Br,L,128)
  const float* oW   = (const float*)d_in[14];  // (Br,L,64,128)
  const float* opW  = (const float*)d_in[15];  // (Br,32,64)
  const float* opB  = (const float*)d_in[16];  // (Br,32)
  float* out = (float*)d_out;

  // workspace partition (floats): H | XZ | U | XDBL | Y | ZE  (~322 MB)
  float* ws = (float*)d_ws;
  size_t o = 0;
  float* Hbuf = ws + o; o += (size_t)BR * MROWS * HDIM;
  float* XZ   = ws + o; o += (size_t)BR * MROWS * (2 * DIN);
  float* Ubuf = ws + o; o += (size_t)BR * MROWS * DIN;
  float* XD   = ws + o; o += (size_t)BR * MROWS * XDBLW;
  float* Ybuf = ws + o; o += (size_t)BR * MROWS * DIN;
  float* ZE   = ws + o; o += (size_t)BR * MROWS * EDIM;

  // embed
  {
    const long tot = (long)BR * MROWS * HDIM;
    embed_kernel<<<dim3((unsigned)((tot + 255) / 256)), 256, 0, stream>>>(
        x0, x1, x2, x3, inpW, inpB, Hbuf);
  }

  for (int l = 0; l < LNUM; ++l) {
    const float* ipW_l = ipW + (long)l * 2 * DIN * HDIM;
    const float* cW_l  = cW  + (long)l * DIN * KCONV;
    const float* cB_l  = cB  + (long)l * DIN;
    const float* xpW_l = xpW + (long)l * XDBLW * DIN;
    const float* dtW_l = dtW + (long)l * DIN * RRANK;
    const float* dtB_l = dtB + (long)l * DIN;
    const float* Al_l  = Alog + (long)l * DIN * NSTATE;
    const float* Dp_l  = Dsk + (long)l * DIN;
    const float* oW_l  = oW  + (long)l * HDIM * DIN;

    // xz = H @ inW^T   (N=256, K=64, 4 n-tiles/wave)
    wmma_gemm<2 * DIN, HDIM, 4, ASrc::Plain, false><<<gemm_grid(2 * DIN, 4), 256, 0, stream>>>(
        Hbuf, (long)MROWS * HDIM, nullptr, 0, 0, 0,
        ipW_l, (long)LNUM * 2 * DIN * HDIM, nullptr, 0,
        XZ, (long)MROWS * 2 * DIN);

    // u = silu(causal depthwise conv(xr) + b)
    conv_silu_kernel<<<dim3((unsigned)(((long)MROWS * DIN + 255) / 256), 1, BR), 256, 0, stream>>>(
        XZ, cW_l, (long)LNUM * DIN * KCONV, cB_l, (long)LNUM * DIN, Ubuf);

    // x_dbl = u @ xpW^T   (N=36 padded to 48, K=128, 3 n-tiles/wave)
    wmma_gemm<XDBLW, DIN, 3, ASrc::Plain, false><<<gemm_grid(XDBLW, 3), 256, 0, stream>>>(
        Ubuf, (long)MROWS * DIN, nullptr, 0, 0, 0,
        xpW_l, (long)LNUM * XDBLW * DIN, nullptr, 0,
        XD, (long)MROWS * XDBLW);

    // selective scan -> y (+ D skip)
    scan_kernel<<<dim3(BATCH, 1, BR), DIN, 0, stream>>>(
        XD, Ubuf,
        dtW_l, (long)LNUM * DIN * RRANK,
        dtB_l, (long)LNUM * DIN,
        Al_l,  (long)LNUM * DIN * NSTATE,
        Dp_l,  (long)LNUM * DIN,
        Ybuf);

    // H = (y * silu(z)) @ outW^T   (N=64, K=128, gate fused, 4 n-tiles/wave)
    wmma_gemm<HDIM, DIN, 4, ASrc::YSiluZ, false><<<gemm_grid(HDIM, 4), 256, 0, stream>>>(
        Ybuf, (long)MROWS * DIN, XZ, (long)MROWS * 2 * DIN, 2 * DIN, DIN,
        oW_l, (long)LNUM * HDIM * DIN, nullptr, 0,
        Hbuf, (long)MROWS * HDIM);
  }

  // zE = tanh(H @ outp_W^T + outp_b)  (N=32, K=64, 2 n-tiles/wave, fused epilogue)
  wmma_gemm<EDIM, HDIM, 2, ASrc::Plain, true><<<gemm_grid(EDIM, 2), 256, 0, stream>>>(
      Hbuf, (long)MROWS * HDIM, nullptr, 0, 0, 0,
      opW, (long)EDIM * HDIM, opB, (long)EDIM,
      ZE, (long)MROWS * EDIM);

  // hyperbolic combine -> 5 stacked outputs
  hyper_kernel<<<dim3((MROWS + 7) / 8), 256, 0, stream>>>(ZE, out);
}
